// RNNDecoder_47717086658819
// MI455X (gfx1250) — matrix operand order repaired and checked by reading, not compiled, not run on hardware
//
#include <hip/hip_runtime.h>
#include <math.h>

typedef __attribute__((ext_vector_type(16))) __bf16        v16bf;
typedef __attribute__((ext_vector_type(8)))  float         v8f;
typedef __attribute__((ext_vector_type(4)))  unsigned int  v4u;

#define B_  64
#define H_  1024
#define V_  32000
#define T_  64
#define LDSLD (H_ + 8)                       // padded LDS row stride (ushorts)
#define LDS_BYTES (B_ * LDSLD * 2)           // 132096 B, fits 2x in 320KB WGP LDS

// ---- d_out layout (floats) ----
#define LOGITS_OFF 0                         // B*T*V = 131072000
#define PRED_OFF   131072000                 // B*T   = 4096
#define HID_OFF    131076096                 // T*B*H = 4194304

// ---- workspace layout (bytes) ----
#define WIH_OFF   0                          // 1024*1024*2  = 2097152
#define WHH_OFF   2097152
#define WOUT_OFF  4194304                    // 32000*1024*2 = 65536000
#define TOK_OFF   69730304                   // 64 ints (pad to 256B)
#define EBF_OFF   69730560                   // 64*1024*2 = 131072
#define HB0_OFF   69861632                   // 131072
#define HB1_OFF   69992704                   // 131072
#define WS_NEED   70123776u

__device__ __forceinline__ unsigned short f2bf(float f) {
    unsigned int u = __float_as_uint(f);
    return (unsigned short)((u + 0x7FFFu + ((u >> 16) & 1u)) >> 16);  // RNE
}

// B fragment (32x16 bf16): lane n (0..15) holds column n, K = k0..k0+7, k0+16..k0+23;
// lanes 16..31 hold K = k0+8..k0+15, k0+24..k0+31. Column n of B == row n of the
// row-major weight matrix (x @ W^T), so loads are contiguous in K.
__device__ __forceinline__ v16bf load_fragB(const unsigned short* w, int ld,
                                            int n0, int k0, int lane) {
    const unsigned short* p = w + (size_t)(n0 + (lane & 15)) * ld
                                + k0 + ((lane >> 4) << 3);
    v16bf b;
    ((v4u*)&b)[0] = *(const v4u*)p;
    ((v4u*)&b)[1] = *(const v4u*)(p + 16);
    return b;
}

// A fragment (16x32 bf16) from a row-major bf16 matrix in global memory:
// lanes 0..15 (row m): K = k0..k0+7 , k0+16..k0+23
// lanes 16..31 (row m): K = k0+8..k0+15 , k0+24..k0+31
__device__ __forceinline__ v16bf load_fragA(const unsigned short* a, int ld,
                                            int m0, int k0, int lane) {
    const unsigned short* p = a + (size_t)(m0 + (lane & 15)) * ld
                                + k0 + ((lane >> 4) << 3);
    v16bf r;
    ((v4u*)&r)[0] = *(const v4u*)p;
    ((v4u*)&r)[1] = *(const v4u*)(p + 16);
    return r;
}

// Same A-fragment gather from the padded LDS image (row stride LDSLD ushorts:
// 2064B = 516 dwords -> successive rows shift 4 banks; the 16 lanes of each
// half-wave hit disjoint 4-dword bank groups => conflict-free ds_load_b128.
__device__ __forceinline__ v16bf load_fragA_lds(const unsigned short* a,
                                                int m0, int k0, int lane) {
    const unsigned short* p = a + (m0 + (lane & 15)) * LDSLD
                                + k0 + ((lane >> 4) << 3);
    v16bf r;
    ((v4u*)&r)[0] = *(const v4u*)p;
    ((v4u*)&r)[1] = *(const v4u*)(p + 16);
    return r;
}

#define WMMA_BF16(A, Bf, C) \
    __builtin_amdgcn_wmma_f32_16x16x32_bf16(false, (A), false, (Bf), (short)0, (C), false, false)

// ---------------- setup kernels ----------------
__global__ void cvt_bf16_kernel(unsigned short* __restrict__ dst,
                                const float* __restrict__ src, int n) {
    int i = blockIdx.x * blockDim.x + threadIdx.x;
    if (i < n) dst[i] = f2bf(src[i]);
}

// h_bf0 = bf16(h0); tok = x
__global__ void init_kernel(const int* __restrict__ x,
                            const float* __restrict__ h0,
                            unsigned short* __restrict__ hb0,
                            int* __restrict__ tok) {
    int i = blockIdx.x * blockDim.x + threadIdx.x;   // 65536
    hb0[i] = f2bf(h0[i]);
    if (i < B_) tok[i] = x[i];
}

// ---------------- per-step kernels ----------------
// e_bf[b][k] = bf16(relu(emb[tok[b]][k]))
__global__ __launch_bounds__(256)
void gather_e_kernel(const float* __restrict__ emb, const int* __restrict__ tok,
                     unsigned short* __restrict__ ebf) {
    int i = blockIdx.x * 256 + threadIdx.x;          // 65536
    int b = i >> 10, k = i & (H_ - 1);
    float v = emb[(size_t)tok[b] * H_ + k];
    ebf[i] = f2bf(fmaxf(v, 0.f));
}

// hn = tanh(e_bf @ W_ih^T + h_bf @ W_hh^T + b_ih + b_hh)
// grid 64 blocks x 128 thr (4 waves); one wave per 16x16 tile; 256 tiles.
// Writes f32 to d_out hiddens slab AND bf16 to ping-pong buffer.
__global__ __launch_bounds__(128)
void hidden_kernel(const unsigned short* __restrict__ ebf,
                   const unsigned short* __restrict__ hbf_in,
                   const unsigned short* __restrict__ Wih,
                   const unsigned short* __restrict__ Whh,
                   const float* __restrict__ bih, const float* __restrict__ bhh,
                   float* __restrict__ hout, unsigned short* __restrict__ hbf_out) {
    int lane = threadIdx.x & 31;
    int tile = blockIdx.x * 4 + (threadIdx.x >> 5);
    int mt   = (tile & 3) * 16;
    int nt   = (tile >> 2) * 16;

    v8f acc = {};
#pragma unroll 4
    for (int k0 = 0; k0 < H_; k0 += 32) {
        v16bf a1 = load_fragA(ebf, H_, mt, k0, lane);
        v16bf b1 = load_fragB(Wih, H_, nt, k0, lane);
        acc = WMMA_BF16(a1, b1, acc);
        v16bf a2 = load_fragA(hbf_in, H_, mt, k0, lane);
        v16bf b2 = load_fragB(Whh, H_, nt, k0, lane);
        acc = WMMA_BF16(a2, b2, acc);
    }

    int n    = nt + (lane & 15);
    float bb = bih[n] + bhh[n];
    int rb   = mt + ((lane >> 4) << 3);
#pragma unroll
    for (int v = 0; v < 8; ++v) {
        int row = rb + v;
        float val = tanhf(acc[v] + bb);
        hout[(size_t)row * H_ + n]    = val;
        hbf_out[(size_t)row * H_ + n] = f2bf(val);
    }
}

// y = hn_bf @ W_out^T + b_out.
//  * hn (128KB bf16) is staged ONCE per workgroup into padded LDS; all 8 waves
//    read A fragments via conflict-free ds_load_b128 (8x less global A traffic).
//  * One wave owns a 64x16 strip (4 M-tiles x 1 vocab tile): the weight
//    fragment is loaded once per K-chunk and reused by 4 WMMAs -> each W_out
//    byte crosses L2->WGP exactly once per step (roofline minimum).
// grid 250 blocks x 256 thr (8 waves) = 2000 waves = 2000 vocab tiles.
__global__ __launch_bounds__(256)
void logits_kernel(const unsigned short* __restrict__ hnbf,
                   const unsigned short* __restrict__ Wout,
                   const float* __restrict__ bout,
                   float* __restrict__ logits, int t) {
    extern __shared__ unsigned short hn_lds[];       // 64 rows x LDSLD ushorts
    int tid  = threadIdx.x;
    int lane = tid & 31;
    int nt   = (blockIdx.x * 8 + (tid >> 5)) * 16;   // vocab tile base

    // cooperative stage: 8192 16-byte chunks (64 rows x 128 chunks)
#pragma unroll
    for (int it = 0; it < 32; ++it) {
        int idx = it * 256 + tid;
        int r = idx >> 7;                            // row 0..63
        int c = idx & 127;                           // 16B chunk in row
        *(v4u*)(hn_lds + r * LDSLD + c * 8) =
            *(const v4u*)(hnbf + r * H_ + c * 8);
    }
    __syncthreads();

    v8f acc0 = {}, acc1 = {}, acc2 = {}, acc3 = {};
#pragma unroll 4
    for (int k0 = 0; k0 < H_; k0 += 32) {
        v16bf b  = load_fragB(Wout, H_, nt, k0, lane);
        v16bf a0 = load_fragA_lds(hn_lds,  0, k0, lane);
        acc0 = WMMA_BF16(a0, b, acc0);
        v16bf a1 = load_fragA_lds(hn_lds, 16, k0, lane);
        acc1 = WMMA_BF16(a1, b, acc1);
        v16bf a2 = load_fragA_lds(hn_lds, 32, k0, lane);
        acc2 = WMMA_BF16(a2, b, acc2);
        v16bf a3 = load_fragA_lds(hn_lds, 48, k0, lane);
        acc3 = WMMA_BF16(a3, b, acc3);
    }

    int n    = nt + (lane & 15);
    float bb = bout[n];
    int rb   = (lane >> 4) << 3;
    v8f accs[4] = {acc0, acc1, acc2, acc3};
#pragma unroll
    for (int mtile = 0; mtile < 4; ++mtile) {
#pragma unroll
        for (int v = 0; v < 8; ++v) {
            int row = mtile * 16 + rb + v;           // batch index
            logits[((size_t)row * T_ + t) * V_ + n] = accs[mtile][v] + bb;
        }
    }
}

// argmax over V per batch row; writes next token + prediction-as-float.
__global__ __launch_bounds__(256)
void argmax_kernel(const float* __restrict__ logits, int t,
                   float* __restrict__ pred, int* __restrict__ tok) {
    __shared__ float sv[256];
    __shared__ int   si[256];
    int b   = blockIdx.x;
    int tid = threadIdx.x;
    const float* row = logits + ((size_t)b * T_ + t) * V_;

    float best = -INFINITY;
    int   bidx = 0;
    for (int v = tid; v < V_; v += 256) {
        float x = row[v];
        if (x > best) { best = x; bidx = v; }   // ascending scan: first max kept
    }
    sv[tid] = best; si[tid] = bidx;
    __syncthreads();
    for (int s = 128; s > 0; s >>= 1) {
        if (tid < s) {
            float ov = sv[tid + s]; int oi = si[tid + s];
            if (ov > sv[tid] || (ov == sv[tid] && oi < si[tid])) {
                sv[tid] = ov; si[tid] = oi;
            }
        }
        __syncthreads();
    }
    if (tid == 0) {
        tok[b] = si[0];
        pred[(size_t)b * T_ + t] = (float)si[0];
    }
}

extern "C" void kernel_launch(void* const* d_in, const int* in_sizes, int n_in,
                              void* d_out, int out_size, void* d_ws, size_t ws_size,
                              hipStream_t stream) {
    (void)in_sizes; (void)n_in; (void)out_size;
    if (ws_size < (size_t)WS_NEED) return;

    const int*   x     = (const int*)  d_in[0];
    const float* h0    = (const float*)d_in[1];   // (1, B, H)
    const float* emb   = (const float*)d_in[3];
    const float* W_ih  = (const float*)d_in[4];
    const float* b_ih  = (const float*)d_in[5];
    const float* W_hh  = (const float*)d_in[6];
    const float* b_hh  = (const float*)d_in[7];
    const float* W_out = (const float*)d_in[8];
    const float* b_out = (const float*)d_in[9];

    char* ws = (char*)d_ws;
    unsigned short* Wih_bf  = (unsigned short*)(ws + WIH_OFF);
    unsigned short* Whh_bf  = (unsigned short*)(ws + WHH_OFF);
    unsigned short* Wout_bf = (unsigned short*)(ws + WOUT_OFF);
    int*            tok     = (int*)(ws + TOK_OFF);
    unsigned short* ebf     = (unsigned short*)(ws + EBF_OFF);
    unsigned short* hb[2]   = {(unsigned short*)(ws + HB0_OFF),
                               (unsigned short*)(ws + HB1_OFF)};

    float* out     = (float*)d_out;
    float* logits  = out + LOGITS_OFF;
    float* pred    = out + PRED_OFF;
    float* hiddens = out + HID_OFF;

    // per-call setup: bf16 weights + initial state
    cvt_bf16_kernel<<<(H_ * H_ + 255) / 256, 256, 0, stream>>>(Wih_bf,  W_ih,  H_ * H_);
    cvt_bf16_kernel<<<(H_ * H_ + 255) / 256, 256, 0, stream>>>(Whh_bf,  W_hh,  H_ * H_);
    cvt_bf16_kernel<<<(V_ * H_ + 255) / 256, 256, 0, stream>>>(Wout_bf, W_out, V_ * H_);
    init_kernel<<<(B_ * H_) / 256, 256, 0, stream>>>(x, h0, hb[0], tok);

    for (int t = 0; t < T_; ++t) {
        const unsigned short* hcur_bf = hb[t & 1];
        unsigned short*       hnxt_bf = hb[(t + 1) & 1];
        float* hcur = hiddens + (size_t)t * B_ * H_;

        gather_e_kernel<<<(B_ * H_) / 256, 256, 0, stream>>>(emb, tok, ebf);
        hidden_kernel<<<64, 128, 0, stream>>>(ebf, hcur_bf, Wih_bf, Whh_bf,
                                              b_ih, b_hh, hcur, hnxt_bf);
        logits_kernel<<<250, 256, LDS_BYTES, stream>>>(hnxt_bf, Wout_bf, b_out,
                                                       logits, t);
        argmax_kernel<<<64, 256, 0, stream>>>(logits, t, pred, tok);
    }
}
